// DeepFMPfd_39659728011365
// MI455X (gfx1250) — matrix operand order repaired and checked by reading, not compiled, not run on hardware
//
#include <hip/hip_runtime.h>
#include <hip/hip_bf16.h>

// ---------------------------------------------------------------------------
// DeepFM student/teacher towers, fused for MI455X (gfx1250, wave32, WMMA).
//  - MLP GEMMs (≈91 GFLOP) run on v_wmma_f32_16x16x32_f16.
//  - Per-64-row tile, all activations (H1..H3) stay in 232KB of LDS
//    (WGP has 320KB), eliminating ~460MB of HBM round trips.
//  - Each wave owns a 16-wide N strip and computes all 4 M-tiles per K-step:
//    1 global B-frag + 4 LDS A-frags -> 4 independent WMMA chains
//    (4x less L2 weight traffic, no serial C-accumulation dependency).
//  - Attention (S=50, 4 heads x 4 dims) is online-softmax in registers.
// ---------------------------------------------------------------------------

typedef _Float16 half_t;
typedef __attribute__((ext_vector_type(16))) _Float16 v16h;
typedef __attribute__((ext_vector_type(8)))  _Float16 v8h;
typedef __attribute__((ext_vector_type(8)))  float    v8f;

static constexpr int BATCH = 32768;
static constexpr int SLATE = 50;

// ------------------------- weight prep (transpose + f16) -------------------
// W is [K][N] f32 row-major; Wt is [N][Kpad] f16, zero-padded for k >= K.
__global__ void prep_weights(const float* __restrict__ W, half_t* __restrict__ Wt,
                             int K, int N, int Kpad) {
  int idx = blockIdx.x * blockDim.x + threadIdx.x;
  int total = N * Kpad;
  if (idx >= total) return;
  int n = idx / Kpad;
  int k = idx - n * Kpad;
  float v = (k < K) ? W[(size_t)k * N + n] : 0.0f;
  Wt[idx] = (half_t)v;
}

// ------------------------- gather + slate attention ------------------------
__device__ __forceinline__ void load16(float* r, const float* __restrict__ p) {
  const float4* q = (const float4*)p;
  float4 a = q[0], b = q[1], c = q[2], d = q[3];
  r[0]=a.x; r[1]=a.y; r[2]=a.z; r[3]=a.w;
  r[4]=b.x; r[5]=b.y; r[6]=b.z; r[7]=b.w;
  r[8]=c.x; r[9]=c.y; r[10]=c.z; r[11]=c.w;
  r[12]=d.x; r[13]=d.y; r[14]=d.z; r[15]=d.w;
}

__global__ void gather_attn(
    const int* __restrict__ users, const int* __restrict__ items,
    const int* __restrict__ slate_ids,
    const float* __restrict__ user_lin, const float* __restrict__ item_lin,
    const float* __restrict__ user_emb, const float* __restrict__ item_emb,
    const float* __restrict__ user_lin_t, const float* __restrict__ item_lin_t,
    const float* __restrict__ user_emb_t, const float* __restrict__ item_emb_t,
    const float* __restrict__ slate_emb,
    const float* __restrict__ qW, const float* __restrict__ qb,
    const float* __restrict__ kW, const float* __restrict__ kb,
    const float* __restrict__ vW, const float* __restrict__ vb,
    half_t* __restrict__ Xs, half_t* __restrict__ Xt,
    float* __restrict__ base_s, float* __restrict__ base_t) {
  __shared__ float sQW[256], sKW[256], sVW[256], sQB[16], sKB[16], sVB[16];
  int tid = threadIdx.x;
  if (tid < 256) { sQW[tid] = qW[tid]; sKW[tid] = kW[tid]; sVW[tid] = vW[tid]; }
  if (tid < 16)  { sQB[tid] = qb[tid]; sKB[tid] = kb[tid]; sVB[tid] = vb[tid]; }
  __syncthreads();

  int b = blockIdx.x * blockDim.x + tid;   // grid sized exactly to BATCH
  int u  = users[b];
  int it = items[b];

  // ---- student tower: dense input + analytic 2-field FM = dot(uE,iE) ----
  float uE[16], iE[16];
  load16(uE, user_emb + (size_t)u * 16);
  load16(iE, item_emb + (size_t)it * 16);
  float fm_s = 0.f;
#pragma unroll
  for (int j = 0; j < 16; ++j) fm_s += uE[j] * iE[j];
  base_s[b] = user_lin[u] + item_lin[it] + fm_s;
#pragma unroll
  for (int j = 0; j < 16; ++j) {
    Xs[(size_t)b * 32 + j]      = (half_t)uE[j];
    Xs[(size_t)b * 32 + 16 + j] = (half_t)iE[j];
  }

  // ---- teacher tower ----
  float uT[16], iT[16], tg[16];
  load16(uT, user_emb_t + (size_t)u * 16);
  load16(iT, item_emb_t + (size_t)it * 16);
  load16(tg, slate_emb + (size_t)it * 16);   // targetS = slate_emb[items]

  float q[16];
#pragma unroll
  for (int j = 0; j < 16; ++j) q[j] = sQB[j];
#pragma unroll
  for (int d = 0; d < 16; ++d) {
    float td = tg[d];
#pragma unroll
    for (int j = 0; j < 16; ++j) q[j] = fmaf(td, sQW[d * 16 + j], q[j]);
  }

  // online softmax over the slate, 4 heads of 4 dims, all in registers
  float m[4] = {-1e30f, -1e30f, -1e30f, -1e30f};
  float l[4] = {0.f, 0.f, 0.f, 0.f};
  float acc[16];
#pragma unroll
  for (int j = 0; j < 16; ++j) acc[j] = 0.f;

#pragma unroll 1
  for (int s = 0; s < SLATE; ++s) {
    int sid = slate_ids[(size_t)b * SLATE + s];
    float e[16];
    load16(e, slate_emb + (size_t)sid * 16);
    float kk[16], vv[16];
#pragma unroll
    for (int j = 0; j < 16; ++j) { kk[j] = sKB[j]; vv[j] = sVB[j]; }
#pragma unroll
    for (int d = 0; d < 16; ++d) {
      float ed = e[d];
#pragma unroll
      for (int j = 0; j < 16; ++j) {
        kk[j] = fmaf(ed, sKW[d * 16 + j], kk[j]);
        vv[j] = fmaf(ed, sVW[d * 16 + j], vv[j]);
      }
    }
#pragma unroll
    for (int h = 0; h < 4; ++h) {
      float sc = 0.f;
#pragma unroll
      for (int d2 = 0; d2 < 4; ++d2) sc += q[h * 4 + d2] * kk[h * 4 + d2];
      sc *= 0.5f;
      float mn = fmaxf(m[h], sc);
      float corr = __expf(m[h] - mn);
      float w = __expf(sc - mn);
      l[h] = l[h] * corr + w;
#pragma unroll
      for (int d2 = 0; d2 < 4; ++d2)
        acc[h * 4 + d2] = acc[h * 4 + d2] * corr + w * vv[h * 4 + d2];
      m[h] = mn;
    }
  }

  float att[16];
#pragma unroll
  for (int j = 0; j < 16; ++j) att[j] = acc[j] / l[j >> 2];

  // 3-field FM = sum of pairwise dots
  float fm_t = 0.f;
#pragma unroll
  for (int j = 0; j < 16; ++j)
    fm_t += uT[j] * iT[j] + uT[j] * att[j] + iT[j] * att[j];
  base_t[b] = user_lin_t[u] + item_lin_t[it] + fm_t;
#pragma unroll
  for (int j = 0; j < 16; ++j) {
    Xt[(size_t)b * 48 + j]      = (half_t)uT[j];
    Xt[(size_t)b * 48 + 16 + j] = (half_t)iT[j];
    Xt[(size_t)b * 48 + 32 + j] = (half_t)att[j];
  }
}

// ------------------------- WMMA fragment loaders ---------------------------
// 16-bit A 16x32 layout (ISA 7.12.2): lane L -> row = L&15, half = L>>4;
// elems 0..7 hold K = half*8 + e, elems 8..15 hold K = 16 + half*8 + (e-8).
__device__ __forceinline__ v16h load_a_frag(const half_t* __restrict__ Hin,
                                            int ldK, int mt, int k0, int lane) {
  int row = mt * 16 + (lane & 15);
  int hf = lane >> 4;
  const v8h* p0 = (const v8h*)(Hin + row * ldK + k0 + hf * 8);
  const v8h* p1 = (const v8h*)(Hin + row * ldK + k0 + 16 + hf * 8);
  v8h lo = *p0, hi = *p1;
  v16h a;
#pragma unroll
  for (int i = 0; i < 8; ++i) { a[i] = lo[i]; a[8 + i] = hi[i]; }
  return a;
}

// B 32x16: lane L -> col n = L&15; elems 0..7 hold K = k0 + 8*(L>>4) + e,
// elems 8..15 hold K = k0 + 16 + 8*(L>>4) + (e-8) (same order as A).
__device__ __forceinline__ v16h load_b_frag(const half_t* __restrict__ Wt,
                                            int ldK, int nt, int k0, int lane) {
  int n = nt * 16 + (lane & 15);
  int kb = k0 + ((lane >> 4) << 3);
  const v8h* p = (const v8h*)(Wt + (size_t)n * ldK + kb);
  v8h lo = p[0], hi = p[2];
  v16h bq;
#pragma unroll
  for (int i = 0; i < 8; ++i) { bq[i] = lo[i]; bq[8 + i] = hi[i]; }
  return bq;
}

// One fused layer: Hout[64][N] = act(Hin[64][K] @ W[K][N] + bias), f16 in LDS.
// Wave w owns N strips nt = w, w+8, ...; per K-step it loads ONE global
// B-fragment and FOUR LDS A-fragments (all M tiles of the 64-row block),
// issuing 4 independent WMMAs -> 4 parallel accumulation chains.
__device__ __forceinline__ void gemm_layer(const half_t* __restrict__ Hin, int K,
                                           const half_t* __restrict__ Wt,
                                           const float* __restrict__ bias,
                                           half_t* __restrict__ Hout, int N,
                                           bool relu, int tid) {
  int wave = tid >> 5;
  int lane = tid & 31;
  int ntiles = N >> 4;
  for (int nt = wave; nt < ntiles; nt += 8) {
    v8f acc0 = {}, acc1 = {}, acc2 = {}, acc3 = {};
    for (int k0 = 0; k0 < K; k0 += 32) {
      v16h bq = load_b_frag(Wt, K, nt, k0, lane);
      v16h a0 = load_a_frag(Hin, K, 0, k0, lane);
      v16h a1 = load_a_frag(Hin, K, 1, k0, lane);
      v16h a2 = load_a_frag(Hin, K, 2, k0, lane);
      v16h a3 = load_a_frag(Hin, K, 3, k0, lane);
      acc0 = __builtin_amdgcn_wmma_f32_16x16x32_f16(false, a0, false, bq, (short)0, acc0, false, false);
      acc1 = __builtin_amdgcn_wmma_f32_16x16x32_f16(false, a1, false, bq, (short)0, acc1, false, false);
      acc2 = __builtin_amdgcn_wmma_f32_16x16x32_f16(false, a2, false, bq, (short)0, acc2, false, false);
      acc3 = __builtin_amdgcn_wmma_f32_16x16x32_f16(false, a3, false, bq, (short)0, acc3, false, false);
    }
    // C/D layout: VGPR r -> M = mt*16 + r + 8*(L>>4), N = nt*16 + (L&15)
    int n = nt * 16 + (lane & 15);
    float bn = bias[n];
    int hf = lane >> 4;
#pragma unroll
    for (int r = 0; r < 8; ++r) {
      int mr = r + 8 * hf;
      float v0 = acc0[r] + bn;
      float v1 = acc1[r] + bn;
      float v2 = acc2[r] + bn;
      float v3 = acc3[r] + bn;
      if (relu) {
        v0 = v0 > 0.f ? v0 : 0.f;
        v1 = v1 > 0.f ? v1 : 0.f;
        v2 = v2 > 0.f ? v2 : 0.f;
        v3 = v3 > 0.f ? v3 : 0.f;
      }
      Hout[(size_t)(mr +  0) * N + n] = (half_t)v0;
      Hout[(size_t)(mr + 16) * N + n] = (half_t)v1;
      Hout[(size_t)(mr + 32) * N + n] = (half_t)v2;
      Hout[(size_t)(mr + 48) * N + n] = (half_t)v3;
    }
  }
}

// ------------------------- fused 4-layer MLP tower -------------------------
// One block = 64 rows; LDS: H0(64x64) H1(64x1024) H2(64x512) H3(64x256) f16.
__global__ void mlp_tower(const half_t* __restrict__ X, int K0, int K0pad,
                          const half_t* __restrict__ W1t, const float* __restrict__ b1,
                          const half_t* __restrict__ W2t, const float* __restrict__ b2,
                          const half_t* __restrict__ W3t, const float* __restrict__ b3,
                          const float* __restrict__ W4, const float* __restrict__ b4,
                          const float* __restrict__ base, float* __restrict__ out) {
  extern __shared__ char smem[];
  half_t* H0 = (half_t*)smem;              // 64*64   =  8KB
  half_t* H1 = H0 + 64 * 64;               // 64*1024 = 128KB
  half_t* H2 = H1 + 64 * 1024;             // 64*512  =  64KB
  half_t* H3 = H2 + 64 * 512;              // 64*256  =  32KB  (232KB total)
  int tid = threadIdx.x;
  int row0 = blockIdx.x * 64;

  for (int i = tid; i < 64 * K0pad; i += 256) {
    int r = i / K0pad;
    int k = i - r * K0pad;
    H0[i] = (k < K0) ? X[(size_t)(row0 + r) * K0 + k] : (half_t)0.f;
  }
  __syncthreads();
  gemm_layer(H0, K0pad, W1t, b1, H1, 1024, true, tid);
  __syncthreads();
  gemm_layer(H1, 1024, W2t, b2, H2, 512, true, tid);
  __syncthreads();
  gemm_layer(H2, 512, W3t, b3, H3, 256, true, tid);
  __syncthreads();

  // layer 4: [64,256] @ [256,1]; 4 lanes per row, shuffle reduce
  int r = tid >> 2;
  int part = tid & 3;
  float s = 0.f;
  int kbeg = part * 64;
#pragma unroll 8
  for (int k = kbeg; k < kbeg + 64; ++k)
    s = fmaf((float)H3[(size_t)r * 256 + k], W4[k], s);
  s += __shfl_xor(s, 1, 32);
  s += __shfl_xor(s, 2, 32);
  if (part == 0) out[row0 + r] = base[row0 + r] + s + b4[0];
}

// ------------------------- outputs + reg loss ------------------------------
__global__ void finalize(const float* __restrict__ outT, const float* __restrict__ outS,
                         float* __restrict__ d_out, float* __restrict__ partial) {
  __shared__ float red[256];
  int i = blockIdx.x * 256 + threadIdx.x;
  float t = outT[i], s = outS[i];
  d_out[i]         = 1.f / (1.f + __expf(-t));
  d_out[BATCH + i] = 1.f / (1.f + __expf(-s));
  float d = s - t;
  red[threadIdx.x] = d * d;
  __syncthreads();
  for (int off = 128; off > 0; off >>= 1) {
    if (threadIdx.x < off) red[threadIdx.x] += red[threadIdx.x + off];
    __syncthreads();
  }
  if (threadIdx.x == 0) partial[blockIdx.x] = red[0];
}

__global__ void reduce_reg(const float* __restrict__ partial, float* __restrict__ d_out, int nb) {
  __shared__ float red[128];
  float s = (threadIdx.x < nb) ? partial[threadIdx.x] : 0.f;
  red[threadIdx.x] = s;
  __syncthreads();
  for (int off = 64; off > 0; off >>= 1) {
    if (threadIdx.x < off) red[threadIdx.x] += red[threadIdx.x + off];
    __syncthreads();
  }
  if (threadIdx.x == 0) d_out[2 * BATCH] = red[0] / (float)BATCH;
}

// ---------------------------------------------------------------------------
extern "C" void kernel_launch(void* const* d_in, const int* in_sizes, int n_in,
                              void* d_out, int out_size, void* d_ws, size_t ws_size,
                              hipStream_t stream) {
  // input order per setup_inputs()
  const int*   users      = (const int*)d_in[0];
  const int*   items      = (const int*)d_in[1];
  const int*   slate_ids  = (const int*)d_in[2];
  const float* user_lin   = (const float*)d_in[3];
  const float* item_lin   = (const float*)d_in[4];
  const float* user_emb   = (const float*)d_in[5];
  const float* item_emb   = (const float*)d_in[6];
  const float* mlp_W[4]   = {(const float*)d_in[7],  (const float*)d_in[9],
                             (const float*)d_in[11], (const float*)d_in[13]};
  const float* mlp_b[4]   = {(const float*)d_in[8],  (const float*)d_in[10],
                             (const float*)d_in[12], (const float*)d_in[14]};
  const float* user_lin_t = (const float*)d_in[15];
  const float* item_lin_t = (const float*)d_in[16];
  const float* user_emb_t = (const float*)d_in[17];
  const float* item_emb_t = (const float*)d_in[18];
  const float* mlpt_W[4]  = {(const float*)d_in[19], (const float*)d_in[21],
                             (const float*)d_in[23], (const float*)d_in[25]};
  const float* mlpt_b[4]  = {(const float*)d_in[20], (const float*)d_in[22],
                             (const float*)d_in[24], (const float*)d_in[26]};
  const float* slate_emb  = (const float*)d_in[27];
  const float* qW = (const float*)d_in[28]; const float* qb = (const float*)d_in[29];
  const float* kW = (const float*)d_in[30]; const float* kb = (const float*)d_in[31];
  const float* vW = (const float*)d_in[32]; const float* vb = (const float*)d_in[33];

  // ---- workspace layout ----
  char* ws = (char*)d_ws;
  size_t off = 0;
  auto alloc = [&](size_t bytes) { char* p = ws + off; off += (bytes + 255) & ~size_t(255); return p; };
  half_t* Xs     = (half_t*)alloc((size_t)BATCH * 32 * sizeof(half_t));
  half_t* Xt     = (half_t*)alloc((size_t)BATCH * 48 * sizeof(half_t));
  float*  base_s = (float*)alloc((size_t)BATCH * sizeof(float));
  float*  base_t = (float*)alloc((size_t)BATCH * sizeof(float));
  float*  outS   = (float*)alloc((size_t)BATCH * sizeof(float));
  float*  outT   = (float*)alloc((size_t)BATCH * sizeof(float));
  half_t* W1ts   = (half_t*)alloc((size_t)1024 * 32  * sizeof(half_t));
  half_t* W2ts   = (half_t*)alloc((size_t)512  * 1024 * sizeof(half_t));
  half_t* W3ts   = (half_t*)alloc((size_t)256  * 512  * sizeof(half_t));
  half_t* W1tt   = (half_t*)alloc((size_t)1024 * 64  * sizeof(half_t));
  half_t* W2tt   = (half_t*)alloc((size_t)512  * 1024 * sizeof(half_t));
  half_t* W3tt   = (half_t*)alloc((size_t)256  * 512  * sizeof(half_t));
  float*  partial = (float*)alloc(128 * sizeof(float));
  (void)ws_size; (void)n_in; (void)in_sizes; (void)out_size;

  // ---- transpose + f16-convert weights (L2-resident, once per launch) ----
  auto prep = [&](const float* W, half_t* Wt, int K, int N, int Kpad) {
    int total = N * Kpad;
    prep_weights<<<(total + 255) / 256, 256, 0, stream>>>(W, Wt, K, N, Kpad);
  };
  prep(mlp_W[0],  W1ts, 32,   1024, 32);
  prep(mlp_W[1],  W2ts, 1024, 512,  1024);
  prep(mlp_W[2],  W3ts, 512,  256,  512);
  prep(mlpt_W[0], W1tt, 48,   1024, 64);   // K=48 zero-padded to 64
  prep(mlpt_W[1], W2tt, 1024, 512,  1024);
  prep(mlpt_W[2], W3tt, 512,  256,  512);

  // ---- gather + attention + FM ----
  gather_attn<<<BATCH / 256, 256, 0, stream>>>(
      users, items, slate_ids, user_lin, item_lin, user_emb, item_emb,
      user_lin_t, item_lin_t, user_emb_t, item_emb_t, slate_emb,
      qW, qb, kW, kb, vW, vb, Xs, Xt, base_s, base_t);

  // ---- fused WMMA MLP towers ----
  constexpr size_t SMEM = (size_t)(64 * 64 + 64 * 1024 + 64 * 512 + 64 * 256) * sizeof(half_t);
  static bool attr_set = false;
  if (!attr_set) {
    (void)hipFuncSetAttribute(reinterpret_cast<const void*>(mlp_tower),
                              hipFuncAttributeMaxDynamicSharedMemorySize, (int)SMEM);
    attr_set = true;
  }
  mlp_tower<<<BATCH / 64, 256, SMEM, stream>>>(
      Xs, 32, 32, W1ts, mlp_b[0], W2ts, mlp_b[1], W3ts, mlp_b[2],
      mlp_W[3], mlp_b[3], base_s, outS);
  mlp_tower<<<BATCH / 64, 256, SMEM, stream>>>(
      Xt, 48, 64, W1tt, mlpt_b[0], W2tt, mlpt_b[1], W3tt, mlpt_b[2],
      mlpt_W[3], mlpt_b[3], base_t, outT);

  // ---- sigmoids + reg loss ----
  finalize<<<BATCH / 256, 256, 0, stream>>>(outT, outS, (float*)d_out, partial);
  reduce_reg<<<1, 128, 0, stream>>>(partial, (float*)d_out, BATCH / 256);
}
